// Gloss_Fusion_Transformer_20383914787378
// MI455X (gfx1250) — matrix operand order repaired and checked by reading, not compiled, not run on hardware
//
#include <hip/hip_runtime.h>

// ---------------------------------------------------------------------------
// Problem constants (match reference)
// ---------------------------------------------------------------------------
#define Bsz  16
#define Mseq 1024
#define Dmod 512
#define Hh   8
#define HSd  64
#define FFd  2048

typedef __attribute__((ext_vector_type(16))) _Float16 v16h;
typedef __attribute__((ext_vector_type(8)))  _Float16 v8h;
typedef __attribute__((ext_vector_type(8)))  float    v8f;

struct v8h2 { v8h lo; v8h hi; };
union Frag16 { v16h v; v8h2 p; };

// ---------------------------------------------------------------------------
// WMMA GEMM: C[R x N] = act(A[R x K] @ W[K x N] + bias) (+ residual)
// A fp32 (converted to f16 while staging to LDS), W pre-converted f16,
// f32 accumulate.
// Block: 256 threads (8 waves). Block tile 128(M) x 128(N), LDS stage K=64
// (two 16x16x32 K-substeps). Wave w: (w&3) -> 32-row group, (w>>2) -> 64-col
// group; each wave computes a 32x64 sub-tile = 2x4 WMMA tiles.
// => 16 v_wmma per barrier pair per wave.
// LDS stride 72 halves (36 dwords): conflict-free b128 lane rotation.
// act: 0=none, 1=relu, 2=qgelu
// ---------------------------------------------------------------------------
__global__ __launch_bounds__(256)
void gemm_wmma_kernel(const float* __restrict__ A, const _Float16* __restrict__ W,
                      const float* __restrict__ bias, const float* __restrict__ resid,
                      float* __restrict__ C, int R, int K, int N, int act)
{
  __shared__ _Float16 lds_a[128 * 72];   // [row][k0..63], stride 72 halves
  __shared__ _Float16 lds_b[128 * 72];   // [col][k0..63], stride 72 halves

  const int tid  = threadIdx.x;
  const int lane = tid & 31;
  const int wv   = tid >> 5;       // 0..7
  const int wm   = wv & 3;         // row group (0..3) -> 32 rows
  const int wn   = wv >> 2;        // col group (0..1) -> 64 cols
  const int row0 = blockIdx.y * 128;
  const int col0 = blockIdx.x * 128;

  const v8f vzero = {0.f,0.f,0.f,0.f,0.f,0.f,0.f,0.f};
  v8f acc[2][4];
#pragma unroll
  for (int i = 0; i < 2; ++i)
#pragma unroll
    for (int j = 0; j < 4; ++j) acc[i][j] = vzero;

  // A-staging: thread -> (row, 32-wide half of the 64-wide K stage)
  const int arow = tid >> 1;
  const int acol = (tid & 1) * 32;
  // B-staging: thread -> (k row 0..63, base 8-wide N chunk; 4 passes cover 128)
  const int bk  = tid & 63;
  const int bn0 = (tid >> 6) * 8;

  // fragment lane decomposition (CDNA5 WMMA layouts)
  const int fr  = lane & 15;            // A row / B col / D col
  const int ka  = (lane >> 4) << 3;     // A: K base 0/8 within substep
  const int kb2 = (lane >> 4) << 3;     // B: K base 0/8 within substep
  const int hi  = lane >> 4;            // D: row offset 0/8

  for (int k0 = 0; k0 < K; k0 += 64) {
    // ---- stage A tile 128x64 (fp32 -> f16) ----
    {
      const float* ga = A + (size_t)(row0 + arow) * K + (k0 + acol);
      _Float16*    la = lds_a + arow * 72 + acol;
#pragma unroll
      for (int j = 0; j < 32; j += 4) {
        float4 f = *(const float4*)(ga + j);
        la[j + 0] = (_Float16)f.x; la[j + 1] = (_Float16)f.y;
        la[j + 2] = (_Float16)f.z; la[j + 3] = (_Float16)f.w;
      }
      if (k0 + 64 < K) __builtin_prefetch(ga + 64, 0, 1);  // global_prefetch_b8
    }
    // ---- stage B tile 64(K) x 128(N), transposed to [n][k] ----
#pragma unroll
    for (int jp = 0; jp < 4; ++jp) {
      const int bn = bn0 + 32 * jp;
      const _Float16* gb = W + (size_t)(k0 + bk) * N + (col0 + bn);
      v8h bv = *(const v8h*)gb;
#pragma unroll
      for (int e = 0; e < 8; ++e)
        lds_b[(bn + e) * 72 + bk] = bv[e];
    }
    __syncthreads();

    // ---- 2 K-substeps x (2x4) WMMA tiles ----
#pragma unroll
    for (int s = 0; s < 2; ++s) {
      const int ks = s * 32;
      Frag16 fa[2], fb[4];
#pragma unroll
      for (int i = 0; i < 2; ++i) {
        const _Float16* pa = lds_a + (wm * 32 + i * 16 + fr) * 72 + ks + ka;
        fa[i].p.lo = *(const v8h*)(pa);         // K = ka..ka+7
        fa[i].p.hi = *(const v8h*)(pa + 16);    // K = ka+16..ka+23
      }
#pragma unroll
      for (int j = 0; j < 4; ++j) {
        const _Float16* pb = lds_b + (wn * 64 + j * 16 + fr) * 72 + ks + kb2;
        fb[j].p.lo = *(const v8h*)(pb);         // K = kb2..kb2+7
        fb[j].p.hi = *(const v8h*)(pb + 16);    // K = kb2+16..kb2+23
      }
#pragma unroll
      for (int i = 0; i < 2; ++i)
#pragma unroll
        for (int j = 0; j < 4; ++j)
          acc[i][j] = __builtin_amdgcn_wmma_f32_16x16x32_f16(
              false, fa[i].v, false, fb[j].v, (short)0, acc[i][j], false, false);
    }
    __syncthreads();
  }

  // ---- epilogue: bias + activation + residual, fp32 store ----
#pragma unroll
  for (int i = 0; i < 2; ++i) {
#pragma unroll
    for (int j = 0; j < 4; ++j) {
      const int col = col0 + wn * 64 + j * 16 + fr;
      const float bv = bias ? bias[col] : 0.f;
#pragma unroll
      for (int e = 0; e < 8; ++e) {
        const int row = row0 + wm * 32 + i * 16 + e + 8 * hi;
        float v = acc[i][j][e] + bv;
        if (act == 1)      v = fmaxf(v, 0.f);
        else if (act == 2) v = v / (1.f + __expf(-1.702f * v));   // qgelu
        if (resid) v += resid[(size_t)row * N + col];
        C[(size_t)row * N + col] = v;
      }
    }
  }
}

// ---------------------------------------------------------------------------
// Small-N GEMM (offset projection, N=40/24): one thread per output element.
// ---------------------------------------------------------------------------
__global__ void gemm_small_kernel(const float* __restrict__ A, const float* __restrict__ W,
                                  const float* __restrict__ bias, float* __restrict__ C,
                                  int R, int K, int N)
{
  const int i = blockIdx.x * blockDim.x + threadIdx.x;
  if (i >= R * N) return;
  const int r = i / N, n = i % N;
  const float* a = A + (size_t)r * K;
  float s = bias[n];
  for (int k = 0; k < K; ++k) s += a[k] * W[(size_t)k * N + n];
  C[i] = s;
}

// ---------------------------------------------------------------------------
// fp32 -> f16 weight conversion
// ---------------------------------------------------------------------------
__global__ void f32_to_f16_kernel(const float* __restrict__ s, _Float16* __restrict__ d, int n)
{
  const int i = blockIdx.x * blockDim.x + threadIdx.x;
  if (i < n) d[i] = (_Float16)s[i];
}

// ---------------------------------------------------------------------------
// pose/rgb + sinusoidal positional encoding
// ---------------------------------------------------------------------------
__global__ void add_pe_kernel(const float* __restrict__ in, float* __restrict__ out, int n)
{
  const int i = blockIdx.x * blockDim.x + threadIdx.x;
  if (i >= n) return;
  const int d = i % Dmod;
  const int m = (i / Dmod) % Mseq;
  const float ang = (float)m * __expf(-(float)(d & ~1) * (logf(10000.f) / (float)Dmod));
  const float pe = (d & 1) ? __cosf(ang) : __sinf(ang);
  out[i] = in[i] + pe;
}

// ---------------------------------------------------------------------------
// sign_lens[b] = sum(1 - mask[b,0,:]) - 1
// ---------------------------------------------------------------------------
__global__ void signlen_kernel(const int* __restrict__ mask, float* __restrict__ sl)
{
  const int b = threadIdx.x;
  if (b >= Bsz) return;
  int s = 0;
  for (int i = 0; i < Mseq; ++i) s += 1 - mask[b * Mseq + i];
  sl[b] = (float)s - 1.f;
}

// ---------------------------------------------------------------------------
// Deformable attention: one wave per (b, m, head); lane owns 2 of 64 dims.
// ---------------------------------------------------------------------------
__global__ __launch_bounds__(256)
void deform_attn_kernel(const float* __restrict__ kbuf, const float* __restrict__ vbuf,
                        const float* __restrict__ qbuf, const float* __restrict__ offb,
                        const float* __restrict__ signlen, float* __restrict__ ctx, int nk)
{
  const int wid  = blockIdx.x * 8 + (threadIdx.x >> 5);
  const int lane = threadIdx.x & 31;
  const int h = wid % Hh;
  const int m = (wid / Hh) % Mseq;
  const int b = wid / (Hh * Mseq);
  const size_t rowq = (size_t)b * Mseq + m;
  const int dbase = h * HSd + lane * 2;

  const float2 qv = *(const float2*)(qbuf + rowq * Dmod + dbase);
  const float L = signlen[b];
  const int hnk = Hh * nk;
  const int left = -((nk + 1) >> 1);           // Python -nk//2 (floor)
  const float Wf = (float)Mseq;

  float sc[5], svx[5], svy[5];

  for (int kk = 0; kk < nk; ++kk) {
    const float offv = offb[rowq * hnk + h * nk + kk];
    float s = (float)(left + kk + m) + offv;
    float r = fmodf(s, L); if (r < 0.f) r += L;            // jnp.mod
    const float xpix = r * (Wf / (Wf - 1.f)) - 0.5f;
    const float x0 = floorf(xpix);
    const float f = xpix - x0;
    const int i0 = (int)x0, i1 = i0 + 1;
    float k0x=0,k0y=0,v0x=0,v0y=0,k1x=0,k1y=0,v1x=0,v1y=0;
    if (i0 >= 0 && i0 < Mseq) {
      const size_t base = ((size_t)b * Mseq + i0) * Dmod + dbase;
      float2 t = *(const float2*)(kbuf + base); k0x = t.x; k0y = t.y;
      t = *(const float2*)(vbuf + base);        v0x = t.x; v0y = t.y;
    }
    if (i1 >= 0 && i1 < Mseq) {
      const size_t base = ((size_t)b * Mseq + i1) * Dmod + dbase;
      float2 t = *(const float2*)(kbuf + base); k1x = t.x; k1y = t.y;
      t = *(const float2*)(vbuf + base);        v1x = t.x; v1y = t.y;
    }
    const float g = 1.f - f;
    const float skx = (g * k0x + f * k1x) * 0.5f;
    const float sky = (g * k0y + f * k1y) * 0.5f;
    svx[kk] = (g * v0x + f * v1x) * 0.5f;
    svy[kk] = (g * v0y + f * v1y) * 0.5f;
    float p = qv.x * skx + qv.y * sky;
#pragma unroll
    for (int t = 16; t >= 1; t >>= 1) p += __shfl_xor(p, t, 32);
    sc[kk] = p * 0.125f;                        // / sqrt(64)
  }
  float mx = sc[0];
  for (int kk = 1; kk < nk; ++kk) mx = fmaxf(mx, sc[kk]);
  float den = 0.f;
  for (int kk = 0; kk < nk; ++kk) { float e = __expf(sc[kk] - mx); sc[kk] = e; den += e; }
  const float inv = 1.f / den;
  float cx = 0.f, cy = 0.f;
  for (int kk = 0; kk < nk; ++kk) { const float a = sc[kk] * inv; cx += a * svx[kk]; cy += a * svy[kk]; }
  *(float2*)(ctx + rowq * Dmod + dbase) = make_float2(cx, cy);
}

// ---------------------------------------------------------------------------
// LayerNorm over D=512: one wave per row (16 elems/lane).
// ---------------------------------------------------------------------------
__global__ __launch_bounds__(256)
void layernorm_kernel(const float* __restrict__ x, const float* __restrict__ g,
                      const float* __restrict__ bt, float* __restrict__ y, float eps)
{
  const int row  = blockIdx.x * 8 + (threadIdx.x >> 5);
  const int lane = threadIdx.x & 31;
  const float* xr = x + (size_t)row * Dmod + lane * 16;
  float v[16];
#pragma unroll
  for (int j = 0; j < 16; j += 4) {
    float4 t = *(const float4*)(xr + j);
    v[j] = t.x; v[j+1] = t.y; v[j+2] = t.z; v[j+3] = t.w;
  }
  float s = 0.f;
#pragma unroll
  for (int j = 0; j < 16; ++j) s += v[j];
#pragma unroll
  for (int t = 16; t >= 1; t >>= 1) s += __shfl_xor(s, t, 32);
  const float mu = s * (1.f / Dmod);
  float q = 0.f;
#pragma unroll
  for (int j = 0; j < 16; ++j) { const float d = v[j] - mu; q += d * d; }
#pragma unroll
  for (int t = 16; t >= 1; t >>= 1) q += __shfl_xor(q, t, 32);
  const float rs = rsqrtf(q * (1.f / Dmod) + eps);
  float* yr = y + (size_t)row * Dmod + lane * 16;
  const float* gr = g + lane * 16;
  const float* br = bt + lane * 16;
#pragma unroll
  for (int j = 0; j < 16; ++j) yr[j] = (v[j] - mu) * rs * gr[j] + br[j];
}

// Final: x = h3 + pose + rgb; LayerNorm(eps=1e-5) -> d_out
__global__ __launch_bounds__(256)
void fuse_final_kernel(const float* __restrict__ h3, const float* __restrict__ pa,
                       const float* __restrict__ pb, const float* __restrict__ g,
                       const float* __restrict__ bt, float* __restrict__ y, float eps)
{
  const int row  = blockIdx.x * 8 + (threadIdx.x >> 5);
  const int lane = threadIdx.x & 31;
  const size_t base = (size_t)row * Dmod + lane * 16;
  float v[16];
#pragma unroll
  for (int j = 0; j < 16; ++j) v[j] = h3[base + j] + pa[base + j] + pb[base + j];
  float s = 0.f;
#pragma unroll
  for (int j = 0; j < 16; ++j) s += v[j];
#pragma unroll
  for (int t = 16; t >= 1; t >>= 1) s += __shfl_xor(s, t, 32);
  const float mu = s * (1.f / Dmod);
  float q = 0.f;
#pragma unroll
  for (int j = 0; j < 16; ++j) { const float d = v[j] - mu; q += d * d; }
#pragma unroll
  for (int t = 16; t >= 1; t >>= 1) q += __shfl_xor(q, t, 32);
  const float rs = rsqrtf(q * (1.f / Dmod) + eps);
  const float* gr = g + lane * 16;
  const float* br = bt + lane * 16;
#pragma unroll
  for (int j = 0; j < 16; ++j) y[base + j] = (v[j] - mu) * rs * gr[j] + br[j];
}

// Concat rows: X[r, 0:512] = a[r], X[r, 512:1024] = b[r]
__global__ void concat_kernel(const float* __restrict__ a, const float* __restrict__ b,
                              float* __restrict__ o, int n)
{
  const int i = blockIdx.x * blockDim.x + threadIdx.x;
  if (i >= n) return;
  const int c = i % (2 * Dmod);
  const int r = i / (2 * Dmod);
  o[i] = (c < Dmod) ? a[(size_t)r * Dmod + c] : b[(size_t)r * Dmod + (c - Dmod)];
}

// ---------------------------------------------------------------------------
// Host orchestration
// ---------------------------------------------------------------------------
struct LayerW {
  const float *ln_g, *ln_b, *ffln_g, *ffln_b;
  const float *kb, *vb, *qb, *ob, *ff1b, *ff2b, *off_b, *off_w;
  _Float16 *kw, *vw, *qw, *ow, *ff1w, *ff2w;
  int nk;
};

extern "C" void kernel_launch(void* const* d_in, const int* in_sizes, int n_in,
                              void* d_out, int out_size, void* d_ws, size_t ws_size,
                              hipStream_t stream)
{
  (void)in_sizes; (void)n_in; (void)out_size; (void)ws_size;
  const int R = Bsz * Mseq;                    // 16384 rows
  const size_t S = (size_t)R * Dmod;           // 8.4M floats
  const size_t F = (size_t)R * FFd;            // 33.5M floats

  // ---- input indices (pytree flatten: sorted dict keys, list order) ----
  // 0 pose, 1 rgb, 2 mask, 3..10 fus{c_fc.b,c_fc.w,c_proj.b,c_proj.w,
  //   c_proj1.b,c_proj1.w,ln_b,ln_g}, then 18-leaf layer blocks at
  // 11 (p2r[0]), 29 (p2r[1]), 47 (r2p[0]), 65 (r2p[1]).
  const float* in_pose = (const float*)d_in[0];
  const float* in_rgb  = (const float*)d_in[1];
  const int*   in_mask = (const int*)d_in[2];
  const float* fus_cfc_b    = (const float*)d_in[3];
  const float* fus_cfc_w    = (const float*)d_in[4];
  const float* fus_cproj_b  = (const float*)d_in[5];
  const float* fus_cproj_w  = (const float*)d_in[6];
  const float* fus_cproj1_b = (const float*)d_in[7];
  const float* fus_cproj1_w = (const float*)d_in[8];
  const float* fus_ln_b     = (const float*)d_in[9];
  const float* fus_ln_g     = (const float*)d_in[10];

  // ---- workspace carve (needs ~580 MB) ----
  float* ws = (float*)d_ws;
  size_t o = 0;
  float* P   = ws + o; o += S;
  float* Rg  = ws + o; o += S;
  float* Pn  = ws + o; o += S;
  float* Rn  = ws + o; o += S;
  float* T0  = ws + o; o += S;   // qn   / fusion X (T0..T1 contiguous)
  float* T1  = ws + o; o += S;   // k
  float* T2  = ws + o; o += S;   // v    / fusion H2 (T2..T3 contiguous)
  float* T3  = ws + o; o += S;   // q
  float* T4  = ws + o; o += S;   // ctx  / fusion h3
  float* T5  = ws + o; o += S;   // h
  float* T6  = ws + o; o += S;   // hn
  float* TF  = ws + o; o += F;   // ff1 out / fusion h1
  float* OFF = ws + o; o += (size_t)R * Hh * 5;
  float* SL  = ws + o; o += 64;
  _Float16* wh = (_Float16*)(ws + o);
  size_t ho = 0;

  auto conv = [&](const float* src, size_t n) -> _Float16* {
    _Float16* dst = wh + ho; ho += (n + 7) & ~(size_t)7;
    f32_to_f16_kernel<<<(unsigned)((n + 255) / 256), 256, 0, stream>>>(src, dst, (int)n);
    return dst;
  };
  auto gemm = [&](const float* Ap, const _Float16* Wp, const float* bp,
                  const float* rp, float* Cp, int K, int N, int act) {
    dim3 grid(N / 128, R / 128);
    gemm_wmma_kernel<<<grid, 256, 0, stream>>>(Ap, Wp, bp, rp, Cp, R, K, N, act);
  };

  // ---- pos-enc, sign lengths ----
  add_pe_kernel<<<(unsigned)((S + 255) / 256), 256, 0, stream>>>(in_pose, P, (int)S);
  add_pe_kernel<<<(unsigned)((S + 255) / 256), 256, 0, stream>>>(in_rgb, Rg, (int)S);
  signlen_kernel<<<1, 32, 0, stream>>>(in_mask, SL);

  // ---- convert weights to f16, build layer tables ----
  LayerW LWs[4];
  const int bases[4] = {11, 29, 47, 65};       // p2r0, p2r1, r2p0, r2p1
  const int nks[4]   = {5, 3, 5, 3};
  for (int li = 0; li < 4; ++li) {
    const int bb = bases[li];
    LayerW& L = LWs[li];
    L.nk     = nks[li];
    L.ff1b   = (const float*)d_in[bb + 0];
    L.ff1w   = conv((const float*)d_in[bb + 1], (size_t)Dmod * FFd);
    L.ff2b   = (const float*)d_in[bb + 2];
    L.ff2w   = conv((const float*)d_in[bb + 3], (size_t)FFd * Dmod);
    L.ffln_b = (const float*)d_in[bb + 4];
    L.ffln_g = (const float*)d_in[bb + 5];
    L.kb     = (const float*)d_in[bb + 6];
    L.kw     = conv((const float*)d_in[bb + 7], (size_t)Dmod * Dmod);
    L.ln_b   = (const float*)d_in[bb + 8];
    L.ln_g   = (const float*)d_in[bb + 9];
    L.off_b  = (const float*)d_in[bb + 10];
    L.off_w  = (const float*)d_in[bb + 11];    // stays fp32 (small-N GEMM)
    L.ob     = (const float*)d_in[bb + 12];
    L.ow     = conv((const float*)d_in[bb + 13], (size_t)Dmod * Dmod);
    L.qb     = (const float*)d_in[bb + 14];
    L.qw     = conv((const float*)d_in[bb + 15], (size_t)Dmod * Dmod);
    L.vb     = (const float*)d_in[bb + 16];
    L.vw     = conv((const float*)d_in[bb + 17], (size_t)Dmod * Dmod);
  }
  _Float16* cfc_w    = conv(fus_cfc_w,    (size_t)(2 * Dmod) * (4 * Dmod));
  _Float16* cproj_w  = conv(fus_cproj_w,  (size_t)(4 * Dmod) * (2 * Dmod));
  _Float16* cproj1_w = conv(fus_cproj1_w, (size_t)(2 * Dmod) * Dmod);

  // ---- encoder layer ----
  auto enc_layer = [&](const float* qin, const float* kvin, const LayerW& L, float* dst) {
    layernorm_kernel<<<R / 8, 256, 0, stream>>>(qin, L.ln_g, L.ln_b, T0, 1e-6f);
    gemm(kvin, L.kw, L.kb, nullptr, T1, Dmod, Dmod, 0);          // k
    gemm(kvin, L.vw, L.vb, nullptr, T2, Dmod, Dmod, 0);          // v
    gemm(T0,   L.qw, L.qb, nullptr, T3, Dmod, Dmod, 0);          // q
    const int hnk = Hh * L.nk;
    gemm_small_kernel<<<(unsigned)(((size_t)R * hnk + 255) / 256), 256, 0, stream>>>(
        T0, L.off_w, L.off_b, OFF, R, Dmod, hnk);                // offsets
    deform_attn_kernel<<<(R * Hh) / 8, 256, 0, stream>>>(T1, T2, T3, OFF, SL, T4, L.nk);
    gemm(T4, L.ow, L.ob, qin, T5, Dmod, Dmod, 0);                // out proj + residual
    layernorm_kernel<<<R / 8, 256, 0, stream>>>(T5, L.ffln_g, L.ffln_b, T6, 1e-6f);
    gemm(T6, L.ff1w, L.ff1b, nullptr, TF, Dmod, FFd, 1);         // ff1 + relu
    gemm(TF, L.ff2w, L.ff2b, T5, dst, FFd, Dmod, 0);             // ff2 + residual
  };

  float* curP = P;  float* curR = Rg;
  float* nxtP = Pn; float* nxtR = Rn;
  for (int i = 0; i < 2; ++i) {
    enc_layer(curP, curR, LWs[i],     nxtP);   // p2r[i]: pose queries rgb
    enc_layer(curR, curP, LWs[2 + i], nxtR);   // r2p[i]: rgb queries pose
    float* t = curP; curP = nxtP; nxtP = t;
    t = curR; curR = nxtR; nxtR = t;
  }

  // ---- fusion MLP + final layernorm ----
  float* X2 = T0;   // 2S contiguous (T0,T1)
  float* H2 = T2;   // 2S contiguous (T2,T3)
  concat_kernel<<<(unsigned)(((size_t)R * 2 * Dmod + 255) / 256), 256, 0, stream>>>(
      curP, curR, X2, R * 2 * Dmod);
  gemm(X2, cfc_w,    fus_cfc_b,    nullptr, TF, 2 * Dmod, 4 * Dmod, 2);  // qgelu
  gemm(TF, cproj_w,  fus_cproj_b,  nullptr, H2, 4 * Dmod, 2 * Dmod, 2);  // qgelu
  gemm(H2, cproj1_w, fus_cproj1_b, nullptr, T4, 2 * Dmod, Dmod, 0);
  fuse_final_kernel<<<R / 8, 256, 0, stream>>>(T4, curP, curR, fus_ln_g, fus_ln_b,
                                               (float*)d_out, 1e-5f);
}
